// ProbSparseAttention_6451040878939
// MI455X (gfx1250) — matrix operand order repaired and checked by reading, not compiled, not run on hardware
//
#include <hip/hip_runtime.h>
#include <math.h>

typedef __attribute__((ext_vector_type(16))) __bf16 v16bf;
typedef __attribute__((ext_vector_type(8)))  __bf16 v8bf;
typedef __attribute__((ext_vector_type(8)))  float  v8f;

union V16 { v16bf v; v8bf h[2]; };

constexpr int B  = 4;
constexpr int S  = 2048;
constexpr int H  = 8;
constexpr int D  = 64;
constexpr int DM = H * D;   // 512
constexpr int U  = 38;      // int(5*ln(2048))
constexpr float SCALE = 0.125f; // 1/sqrt(64)

static __device__ __forceinline__ v8f vzero() {
  v8f z;
#pragma unroll
  for (int i = 0; i < 8; ++i) z[i] = 0.0f;
  return z;
}

static __device__ __forceinline__ v8f wmma_bf16(v16bf a, v16bf b, v8f c) {
  // D = A(16x32 bf16) * B(32x16 bf16) + C(16x16 f32)
  return __builtin_amdgcn_wmma_f32_16x16x32_bf16(false, a, false, b, (short)0, c,
                                                 false, false);
}

// ---------------------------------------------------------------------------
// Elementwise conversion kernels
// ---------------------------------------------------------------------------
__global__ void convert_f32_bf16(const float* __restrict__ in,
                                 __bf16* __restrict__ out, int n) {
  for (int i = blockIdx.x * blockDim.x + threadIdx.x; i < n;
       i += gridDim.x * blockDim.x)
    out[i] = (__bf16)in[i];
}

// W[K][N] f32 -> Wt[N][K] bf16 (transposed so B-fragments are contiguous in K)
__global__ void convert_weight_T(const float* __restrict__ W,
                                 __bf16* __restrict__ Wt) {
  for (int i = blockIdx.x * blockDim.x + threadIdx.x; i < DM * DM;
       i += gridDim.x * blockDim.x) {
    int k = i / DM, n = i % DM;
    Wt[(size_t)n * DM + k] = (__bf16)W[i];
  }
}

// v (B,S,DM) bf16 -> vT (B,H,D,S) bf16
__global__ void transpose_v(const __bf16* __restrict__ v,
                            __bf16* __restrict__ vT) {
  for (int i = blockIdx.x * blockDim.x + threadIdx.x; i < B * S * DM;
       i += gridDim.x * blockDim.x) {
    int b = i / (S * DM);
    int r = i % (S * DM);
    int s = r / DM;
    int hd = r % DM;
    int h = hd / D, d = hd % D;
    vT[(((size_t)(b * H + h) * D + d) * S) + s] = v[i];
  }
}

// mean over S of vT rows -> meanv[bh*D + d]
__global__ void meanv_kernel(const __bf16* __restrict__ vT,
                             float* __restrict__ meanv) {
  int t = blockIdx.x * blockDim.x + threadIdx.x;
  if (t >= B * H * D) return;
  const __bf16* r = vT + (size_t)t * S;
  float s = 0.0f;
  for (int i = 0; i < S; ++i) s += (float)r[i];
  meanv[t] = s * (1.0f / (float)S);
}

// ---------------------------------------------------------------------------
// WMMA GEMM: out[M][N] = A[M][K](bf16) * Wt[N][K]^T(bf16) + bias[N]
// block = 128 threads (4 waves), wave tile = 16 rows x 64 cols
// ---------------------------------------------------------------------------
template <bool OUTF32>
__global__ __launch_bounds__(128) void gemm_wmma(
    const __bf16* __restrict__ A, const __bf16* __restrict__ Wt,
    const float* __restrict__ bias, void* __restrict__ outp, int M, int N,
    int K) {
  const int lane = threadIdx.x & 31;
  const int wave = threadIdx.x >> 5;
  const int ln = lane & 15, hg = lane >> 4;
  const int m0 = blockIdx.y * 64 + wave * 16;
  const int n0 = blockIdx.x * 64;

  v8f acc[4];
#pragma unroll
  for (int t = 0; t < 4; ++t) acc[t] = vzero();

  const __bf16* arow = A + (size_t)(m0 + ln) * K;
  for (int k0 = 0; k0 < K; k0 += 32) {
    V16 af;  // A fragment, 16x32, rows = m0..m0+15
    af.h[0] = *(const v8bf*)(arow + k0 + 16 * hg);
    af.h[1] = *(const v8bf*)(arow + k0 + 16 * hg + 8);
#pragma unroll
    for (int t = 0; t < 4; ++t) {
      const __bf16* wrow = Wt + (size_t)(n0 + t * 16 + ln) * K + k0 + 16 * hg;
      v16bf bf = *(const v16bf*)wrow;  // B fragment, K contiguous
      acc[t] = wmma_bf16(af.v, bf, acc[t]);
    }
  }

#pragma unroll
  for (int t = 0; t < 4; ++t) {
    int col = n0 + t * 16 + ln;
    float bv = bias[col];
#pragma unroll
    for (int p = 0; p < 8; ++p) {
      int row = m0 + 8 * hg + p;
      float val = acc[t][p] + bv;
      if (OUTF32)
        ((float*)outp)[(size_t)row * N + col] = val;
      else
        ((__bf16*)outp)[(size_t)row * N + col] = (__bf16)val;
    }
  }
}

// ---------------------------------------------------------------------------
// Sampled-score sparsity: sparsity[bh][s] = max_u(score) - mean_u(score)
// ---------------------------------------------------------------------------
__global__ void sparsity_kernel(const __bf16* __restrict__ qb,
                                const __bf16* __restrict__ kb,
                                const int* __restrict__ idx,
                                float* __restrict__ spars) {
  int t = blockIdx.x * blockDim.x + threadIdx.x;
  if (t >= B * H * S) return;
  int bh = t / S, s = t % S;
  int b = bh / H, h = bh % H;
  const __bf16* qr = qb + (size_t)(b * S + s) * DM + h * D;
  float q[D];
#pragma unroll
  for (int i = 0; i < D; ++i) q[i] = (float)qr[i];
  float mx = -INFINITY, sm = 0.0f;
  for (int u = 0; u < U; ++u) {
    int key = idx[u];
    const __bf16* kr = kb + (size_t)(b * S + key) * DM + h * D;
    float d = 0.0f;
#pragma unroll
    for (int i = 0; i < D; ++i) d += q[i] * (float)kr[i];
    d *= SCALE;
    mx = fmaxf(mx, d);
    sm += d;
  }
  spars[t] = mx - sm * (1.0f / (float)U);
}

// Exact U-th largest per (b,h): iterative argmax with removal (handles ties
// the same way as "sparsity >= min(top_U)").
__global__ void topk_threshold(const float* __restrict__ spars,
                               float* __restrict__ thr) {
  const int bh = blockIdx.x;
  const int tid = threadIdx.x;  // 256 threads
  __shared__ float sv[S];
  __shared__ float redv[256];
  __shared__ int redi[256];
  for (int i = tid; i < S; i += 256) sv[i] = spars[(size_t)bh * S + i];
  __syncthreads();
  for (int it = 0; it < U; ++it) {
    float bv = -INFINITY;
    int bi = 0;
    for (int i = tid; i < S; i += 256) {
      float x = sv[i];
      if (x > bv) { bv = x; bi = i; }
    }
    redv[tid] = bv;
    redi[tid] = bi;
    __syncthreads();
    for (int off = 128; off > 0; off >>= 1) {
      if (tid < off) {
        if (redv[tid + off] > redv[tid] ||
            (redv[tid + off] == redv[tid] && redi[tid + off] < redi[tid])) {
          redv[tid] = redv[tid + off];
          redi[tid] = redi[tid + off];
        }
      }
      __syncthreads();
    }
    if (tid == 0) {
      sv[redi[0]] = -INFINITY;
      if (it == U - 1) thr[bh] = redv[0];
    }
    __syncthreads();
  }
}

// ---------------------------------------------------------------------------
// Flash-style full attention + ProbSparse select.
// grid = (B*H, S/64), block = 128 (4 waves), wave = 16 query rows.
// Keys processed 64 per chunk: 8 QK WMMAs + 8 PV WMMAs per softmax pass,
// halving the cross-lane reduction (ds_bpermute) cost per key vs 32-chunks.
// ---------------------------------------------------------------------------
__global__ __launch_bounds__(128) void attention_kernel(
    const __bf16* __restrict__ qb, const __bf16* __restrict__ kb,
    const __bf16* __restrict__ vT, const float* __restrict__ spars,
    const float* __restrict__ thr, const float* __restrict__ meanv,
    __bf16* __restrict__ ctx) {
  const int bh = blockIdx.x;
  const int b = bh / H, h = bh % H;
  const int wave = threadIdx.x >> 5;
  const int lane = threadIdx.x & 31;
  const int ln = lane & 15, hg = lane >> 4;
  const int m0 = blockIdx.y * 64 + wave * 16;

  __shared__ __align__(64) __bf16 plds[4][16][64];  // per-wave P staging (2KB)

  // Q fragments (rows m0..m0+15, d split 0..31 / 32..63)
  const __bf16* qrow = qb + (size_t)(b * S + m0 + ln) * DM + h * D;
  V16 q0, q1;
  q0.h[0] = *(const v8bf*)(qrow + 16 * hg);
  q0.h[1] = *(const v8bf*)(qrow + 16 * hg + 8);
  q1.h[0] = *(const v8bf*)(qrow + 32 + 16 * hg);
  q1.h[1] = *(const v8bf*)(qrow + 40 + 16 * hg);

  v8f acc[4];
#pragma unroll
  for (int t = 0; t < 4; ++t) acc[t] = vzero();
  float rmax[8], rsum[8];
#pragma unroll
  for (int p = 0; p < 8; ++p) { rmax[p] = -INFINITY; rsum[p] = 0.0f; }

  const size_t kbase = (size_t)(b * S) * DM + h * D;
  const size_t vbase = (size_t)bh * D * S;

  for (int kc = 0; kc < S; kc += 64) {
    // Scores for 4 key tiles (16 keys each), K-dim = d in two 32-halves.
    v8f sc[4];
#pragma unroll
    for (int j = 0; j < 4; ++j) {
      const __bf16* kr =
          kb + kbase + (size_t)(kc + j * 16 + ln) * DM + 16 * hg;
      v16bf bk0 = *(const v16bf*)(kr);        // d = 0..31 slice
      v16bf bk1 = *(const v16bf*)(kr + 32);   // d = 32..63 slice
      sc[j] = wmma_bf16(q0.v, bk0, vzero());
      sc[j] = wmma_bf16(q1.v, bk1, sc[j]);
    }

    // Online softmax over the 64-key chunk.
    float pv[4][8], al[8];
#pragma unroll
    for (int p = 0; p < 8; ++p) {
      float sj[4];
#pragma unroll
      for (int j = 0; j < 4; ++j) sj[j] = sc[j][p] * SCALE;
      float m = fmaxf(fmaxf(sj[0], sj[1]), fmaxf(sj[2], sj[3]));
#pragma unroll
      for (int msk = 1; msk < 16; msk <<= 1)
        m = fmaxf(m, __shfl_xor(m, msk, 32));
      float nm = fmaxf(rmax[p], m);
      al[p] = __expf(rmax[p] - nm);
      rmax[p] = nm;
      float ps = 0.0f;
#pragma unroll
      for (int j = 0; j < 4; ++j) {
        pv[j][p] = __expf(sj[j] - nm);
        ps += pv[j][p];
      }
#pragma unroll
      for (int msk = 1; msk < 16; msk <<= 1) ps += __shfl_xor(ps, msk, 32);
      rsum[p] = rsum[p] * al[p] + ps;
    }
#pragma unroll
    for (int t = 0; t < 4; ++t)
#pragma unroll
      for (int p = 0; p < 8; ++p) acc[t][p] *= al[p];

    // Stage P (16x64, C-layout) into LDS, reload as two 16x32 A-fragments.
    asm volatile("" ::: "memory");
#pragma unroll
    for (int j = 0; j < 4; ++j)
#pragma unroll
      for (int p = 0; p < 8; ++p)
        plds[wave][8 * hg + p][j * 16 + ln] = (__bf16)pv[j][p];
    asm volatile("s_wait_dscnt 0" ::: "memory");  // DS in-order within wave
    V16 pf0, pf1;
    pf0.h[0] = *(const v8bf*)&plds[wave][ln][16 * hg];
    pf0.h[1] = *(const v8bf*)&plds[wave][ln][16 * hg + 8];
    pf1.h[0] = *(const v8bf*)&plds[wave][ln][32 + 16 * hg];
    pf1.h[1] = *(const v8bf*)&plds[wave][ln][40 + 16 * hg];
    asm volatile("" ::: "memory");

    // PV: B fragments from vT (keys contiguous along S).
#pragma unroll
    for (int t = 0; t < 4; ++t) {
      const __bf16* vr = vT + vbase + (size_t)(t * 16 + ln) * S + kc + 16 * hg;
      v16bf bv0 = *(const v16bf*)(vr);        // keys kc+0..31 slice
      v16bf bv1 = *(const v16bf*)(vr + 32);   // keys kc+32..63 slice
      acc[t] = wmma_bf16(pf0.v, bv0, acc[t]);
      acc[t] = wmma_bf16(pf1.v, bv1, acc[t]);
    }
  }

  // Epilogue: normalize, select active vs mean_v, write context (bf16).
  float inv[8];
  bool act[8];
  float th = thr[bh];
#pragma unroll
  for (int p = 0; p < 8; ++p) {
    inv[p] = 1.0f / rsum[p];
    int row = m0 + 8 * hg + p;
    act[p] = spars[(size_t)bh * S + row] >= th;
  }
#pragma unroll
  for (int t = 0; t < 4; ++t) {
    int d = t * 16 + ln;
    float mv = meanv[bh * D + d];
#pragma unroll
    for (int p = 0; p < 8; ++p) {
      int row = m0 + 8 * hg + p;
      float val = act[p] ? acc[t][p] * inv[p] : mv;
      ctx[(size_t)(b * S + row) * DM + h * D + d] = (__bf16)val;
    }
  }
}

// ---------------------------------------------------------------------------
extern "C" void kernel_launch(void* const* d_in, const int* in_sizes, int n_in,
                              void* d_out, int out_size, void* d_ws,
                              size_t ws_size, hipStream_t stream) {
  (void)in_sizes; (void)n_in; (void)out_size; (void)ws_size;

  const float* query = (const float*)d_in[0];
  const float* value = (const float*)d_in[1];
  const float* Wq = (const float*)d_in[2];
  const float* bq = (const float*)d_in[3];
  const float* Wk = (const float*)d_in[4];
  const float* bk = (const float*)d_in[5];
  const float* Wv = (const float*)d_in[6];
  const float* bv = (const float*)d_in[7];
  const float* Wo = (const float*)d_in[8];
  const float* bo = (const float*)d_in[9];
  const int*   idx = (const int*)d_in[10];
  float* out = (float*)d_out;

  const size_t eAct = (size_t)B * S * DM;       // activation elements
  const size_t bAct = eAct * sizeof(__bf16);    // 8 MB
  const size_t bW = (size_t)DM * DM * sizeof(__bf16);  // 512 KB

  char* p = (char*)d_ws;
  __bf16* qin = (__bf16*)p; p += bAct;
  __bf16* vin = (__bf16*)p; p += bAct;
  __bf16* WqT = (__bf16*)p; p += bW;
  __bf16* WkT = (__bf16*)p; p += bW;
  __bf16* WvT = (__bf16*)p; p += bW;
  __bf16* WoT = (__bf16*)p; p += bW;
  __bf16* qbuf = (__bf16*)p; p += bAct;
  __bf16* kbuf = (__bf16*)p; p += bAct;
  __bf16* vbuf = (__bf16*)p; p += bAct;
  __bf16* vT   = (__bf16*)p; p += bAct;
  __bf16* ctx  = (__bf16*)p; p += bAct;
  float* spars = (float*)p; p += (size_t)B * H * S * sizeof(float);
  float* thr   = (float*)p; p += 256;
  float* mv    = (float*)p; p += (size_t)B * H * D * sizeof(float);

  // 1. conversions
  convert_f32_bf16<<<4096, 256, 0, stream>>>(query, qin, (int)eAct);
  convert_f32_bf16<<<4096, 256, 0, stream>>>(value, vin, (int)eAct);
  convert_weight_T<<<512, 256, 0, stream>>>(Wq, WqT);
  convert_weight_T<<<512, 256, 0, stream>>>(Wk, WkT);
  convert_weight_T<<<512, 256, 0, stream>>>(Wv, WvT);
  convert_weight_T<<<512, 256, 0, stream>>>(Wo, WoT);

  // 2. projections (bf16 out)
  dim3 ggrid(DM / 64, (B * S) / 64);
  gemm_wmma<false><<<ggrid, 128, 0, stream>>>(qin, WqT, bq, qbuf, B * S, DM, DM);
  gemm_wmma<false><<<ggrid, 128, 0, stream>>>(vin, WkT, bk, kbuf, B * S, DM, DM);
  gemm_wmma<false><<<ggrid, 128, 0, stream>>>(vin, WvT, bv, vbuf, B * S, DM, DM);

  // 3. V transpose + mean_v
  transpose_v<<<4096, 256, 0, stream>>>(vbuf, vT);
  meanv_kernel<<<(B * H * D + 255) / 256, 256, 0, stream>>>(vT, mv);

  // 4. sampled sparsity + threshold
  sparsity_kernel<<<(B * H * S + 255) / 256, 256, 0, stream>>>(qbuf, kbuf, idx,
                                                               spars);
  topk_threshold<<<B * H, 256, 0, stream>>>(spars, thr);

  // 5. attention with ProbSparse select
  attention_kernel<<<dim3(B * H, S / 64), 128, 0, stream>>>(qbuf, kbuf, vT,
                                                            spars, thr, mv, ctx);

  // 6. output projection (f32 out)
  gemm_wmma<true><<<ggrid, 128, 0, stream>>>(ctx, WoT, bo, out, B * S, DM, DM);
}
